// MonetMoVDE_8770323219139
// MI455X (gfx1250) — matrix operand order repaired and checked by reading, not compiled, not run on hardware
//
#include <hip/hip_runtime.h>
#include <hip/hip_bf16.h>
#include <stdint.h>

// Problem constants (from reference)
#define N_TOK  8192      // B*S
#define D_HID  1024
#define E_EXP  512
#define M_DIM  6
#define H_HEAD 6
#define F_FLAT 3072      // E*M == H*E (both 3072)
#define D_HALF 512

typedef __attribute__((ext_vector_type(16))) __bf16 v16bf;
typedef __attribute__((ext_vector_type(8)))  float  v8f;
typedef unsigned short ushort_t;

// float -> bf16 (round-to-nearest-even), raw u16
__device__ __forceinline__ unsigned f2bf(float f) {
    unsigned u = __builtin_bit_cast(unsigned, f);
    u += 0x7FFFu + ((u >> 16) & 1u);
    return u >> 16;
}
// bf16 (raw u16) -> float
__device__ __forceinline__ float bf2f(ushort_t u) {
    return __builtin_bit_cast(float, (unsigned)u << 16);
}

union Frag16 { v16bf v; unsigned u[8]; };

// ---------------------------------------------------------------------------
// C (= or +=) epi( A[N,K]_bf16 @ W[Ncols,K]_bf16^T + bias_f32 )
// Block: 256 thr = 8 waves; block tile 128x128; wave tile 32x64 (2x4 WMMA).
// Explicit LDS double buffer; staging uses four scalar uint4 registers with
// straight-line code (no arrays/lambdas -> guaranteed SROA, no scratch
// alloca): loads for tile k+1 issue before the WMMA pack, LDS stores after.
// ---------------------------------------------------------------------------
template<bool RELU2, bool ACCUM, bool OUT_BF16>
__global__ __launch_bounds__(256)
__attribute__((amdgpu_waves_per_eu(1, 2)))
void gemm_bf16_wmma(
    const ushort_t* __restrict__ A, int lda,
    const ushort_t* __restrict__ W, int ldw,
    const float* __restrict__ bias,
    void* __restrict__ Cout, int ldc, int colOff, int K)
{
    // [2 buffers][128 rows][40 bf16] ; 40 = 32 + 8 pad (80B row stride)
    __shared__ ushort_t As[2 * 128 * 40];
    __shared__ ushort_t Bs[2 * 128 * 40];
    const int BUFSZ = 128 * 40;

    const int tid  = threadIdx.x;
    const int lane = tid & 31;
    const int wave = tid >> 5;
    const int wr   = wave >> 1;      // 0..3 (row wave)
    const int wc   = wave & 1;       // 0..1 (col wave)
    const int kh   = lane >> 4;      // lane half-group
    const int l16  = lane & 15;

    const int rowBase = blockIdx.y * 128;
    const int colBase = blockIdx.x * 128;

    v8f acc[2][4];
    #pragma unroll
    for (int i = 0; i < 2; ++i)
        #pragma unroll
        for (int j = 0; j < 4; ++j)
            #pragma unroll
            for (int v = 0; v < 8; ++v) acc[i][j][v] = 0.0f;

    // Staging geometry: each 128x32 bf16 tile = 512 uint4 chunks; 2 per thread.
    const int row0 = tid >> 2;            // chunk 0: rows 0..63
    const int c80  = (tid & 3) * 8;
    const int row1 = (tid + 256) >> 2;    // chunk 1: rows 64..127
    const int c81  = c80;                 // same column phase

    const ushort_t* Ap0 = A + (size_t)(rowBase + row0) * lda + c80;
    const ushort_t* Ap1 = A + (size_t)(rowBase + row1) * lda + c81;
    const ushort_t* Wp0 = W + (size_t)(colBase + row0) * ldw + c80;
    const ushort_t* Wp1 = W + (size_t)(colBase + row1) * ldw + c81;

    const int sOffA0 = row0 * 40 + c80;
    const int sOffA1 = row1 * 40 + c81;

    // prologue: tile 0 straight into buffer 0 (fused copy, no staging vars)
    *(uint4*)&As[sOffA0] = *(const uint4*)Ap0;
    *(uint4*)&As[sOffA1] = *(const uint4*)Ap1;
    *(uint4*)&Bs[sOffA0] = *(const uint4*)Wp0;
    *(uint4*)&Bs[sOffA1] = *(const uint4*)Wp1;
    __syncthreads();

    int buf = 0;
    for (int k0 = 0; k0 < K; k0 += 32) {
        const bool hasNext = (k0 + 32 < K);

        // issue next tile's global loads (plain scalars -> stay in VGPRs)
        uint4 a0, a1, b0, b1;
        if (hasNext) {
            a0 = *(const uint4*)(Ap0 + k0 + 32);
            a1 = *(const uint4*)(Ap1 + k0 + 32);
            b0 = *(const uint4*)(Wp0 + k0 + 32);
            b1 = *(const uint4*)(Wp1 + k0 + 32);
        }
        if (k0 + 64 < K) {                              // L2 prefetch 2 tiles ahead
            __builtin_prefetch(Ap0 + k0 + 64, 0, 0);
            __builtin_prefetch(Wp0 + k0 + 64, 0, 0);
        }

        // ---- fragments per ISA 7.12.2 lane layouts -------------------------
        const ushort_t* AsB = &As[buf * BUFSZ];
        const ushort_t* BsB = &Bs[buf * BUFSZ];

        // Both A fragments up front (each feeds 4 WMMAs)...
        Frag16 a[2];
        #pragma unroll
        for (int i = 0; i < 2; ++i) {
            int arow = wr * 32 + i * 16 + l16;          // M = lane&15
            const unsigned* p0 = (const unsigned*)(AsB + arow * 40 + 8 * kh);
            const unsigned* p1 = (const unsigned*)(AsB + arow * 40 + 16 + 8 * kh);
            #pragma unroll
            for (int q = 0; q < 4; ++q) { a[i].u[q] = p0[q]; a[i].u[4 + q] = p1[q]; }
        }
        // ...but only one B fragment live at a time (j-outer / i-inner).
        #pragma unroll
        for (int j = 0; j < 4; ++j) {
            Frag16 bj;
            int brow = wc * 64 + j * 16 + l16;          // N = lane&15
            const unsigned* p = (const unsigned*)(BsB + brow * 40 + 8 * kh);
            const unsigned* pb = (const unsigned*)(BsB + brow * 40 + 16 + 8 * kh);
            #pragma unroll
            for (int q = 0; q < 4; ++q) { bj.u[q] = p[q]; bj.u[4 + q] = pb[q]; }
            #pragma unroll
            for (int i = 0; i < 2; ++i)
                acc[i][j] = __builtin_amdgcn_wmma_f32_16x16x32_bf16(
                    false, a[i].v, false, bj.v, short(0), acc[i][j], false, false);
        }

        // store next tile into the opposite buffer (disjoint: no WAR)
        if (hasNext) {
            ushort_t* AsN = &As[(buf ^ 1) * BUFSZ];
            ushort_t* BsN = &Bs[(buf ^ 1) * BUFSZ];
            *(uint4*)&AsN[sOffA0] = a0;
            *(uint4*)&AsN[sOffA1] = a1;
            *(uint4*)&BsN[sOffA0] = b0;
            *(uint4*)&BsN[sOffA1] = b1;
        }
        __syncthreads();
        buf ^= 1;
    }

    // ---- epilogue: D layout — lane: n = lane&15, VGPR v: m = v + 8*kh -------
    #pragma unroll
    for (int i = 0; i < 2; ++i) {
        #pragma unroll
        for (int j = 0; j < 4; ++j) {
            int gcol = colBase + wc * 64 + j * 16 + l16;
            float bv = bias ? bias[gcol] : 0.0f;
            #pragma unroll
            for (int v = 0; v < 8; ++v) {
                int grow = rowBase + wr * 32 + i * 16 + v + 8 * kh;
                float val = acc[i][j][v] + bv;
                if (RELU2) { float t = fmaxf(val, 0.0f); val = t * t; }
                if (OUT_BF16) {
                    ushort_t* p = (ushort_t*)Cout + (size_t)grow * ldc + colOff + gcol;
                    *p = (ushort_t)f2bf(val);
                } else {
                    float* p = (float*)Cout + (size_t)grow * ldc + colOff + gcol;
                    if (ACCUM) *p += val; else *p = val;
                }
            }
        }
    }
}

// ---------------------------------------------------------------------------
// Conversion / elementwise / reduction kernels
// ---------------------------------------------------------------------------

// dst = bf16(src), 8 elements per thread (n8 = n/8 chunks)
__global__ void f32_to_bf16_kernel(const float* __restrict__ src,
                                   ushort_t* __restrict__ dst, int n8) {
    int idx = blockIdx.x * blockDim.x + threadIdx.x;
    if (idx >= n8) return;
    const float4* s = (const float4*)src + (size_t)idx * 2;
    float4 lo = s[0], hi = s[1];
    uint4 o;
    o.x = f2bf(lo.x) | (f2bf(lo.y) << 16);
    o.y = f2bf(lo.z) | (f2bf(lo.w) << 16);
    o.z = f2bf(hi.x) | (f2bf(hi.y) << 16);
    o.w = f2bf(hi.z) | (f2bf(hi.w) << 16);
    ((uint4*)dst)[idx] = o;
}

// G[n,i] = bf16( sum_h g[n,h,i] )
__global__ void gate_sum_kernel(const float* __restrict__ g, ushort_t* __restrict__ G) {
    int idx = blockIdx.x * blockDim.x + threadIdx.x;      // n*512 + i
    if (idx >= N_TOK * E_EXP) return;
    int n = idx >> 9;
    int i = idx & 511;
    const float* gp = g + (size_t)n * F_FLAT + i;
    float s = 0.0f;
    #pragma unroll
    for (int h = 0; h < H_HEAD; ++h) s += gp[h * E_EXP];
    G[idx] = (ushort_t)f2bf(s);
}

// t[n,h,m] = sum_i g[n,h,i] * Y_bf[n, i*M + m]   (t stays f32)
__global__ void head_mix_kernel(const float* __restrict__ g, const ushort_t* __restrict__ Y,
                                float* __restrict__ t) {
    int idx = blockIdx.x * blockDim.x + threadIdx.x;      // n*36 + h*6 + m
    if (idx >= N_TOK * H_HEAD * M_DIM) return;
    int n  = idx / (H_HEAD * M_DIM);
    int hm = idx - n * (H_HEAD * M_DIM);
    int h  = hm / M_DIM;
    int m  = hm - h * M_DIM;
    const float* gp = g + (size_t)n * F_FLAT + h * E_EXP;
    const ushort_t* yp = Y + (size_t)n * F_FLAT + m;
    float s = 0.0f;
    for (int i = 0; i < E_EXP; ++i) s += gp[i] * bf2f(yp[i * M_DIM]);
    t[idx] = s;
}

// In place: Y_bf[n, i*M+m] *= G_bf[n, i]
__global__ void gate_scale_kernel(ushort_t* __restrict__ Y, const ushort_t* __restrict__ G) {
    int idx = blockIdx.x * blockDim.x + threadIdx.x;      // n*F + f
    if (idx >= N_TOK * F_FLAT) return;
    int n = idx / F_FLAT;
    int f = idx - n * F_FLAT;
    Y[idx] = (ushort_t)f2bf(bf2f(Y[idx]) * bf2f(G[n * E_EXP + f / M_DIM]));
}

// X_bf[n, i*M+m] = bf16( sum_h t[n,h,m] * g[n,h,i] )
__global__ void cross_expand_kernel(const float* __restrict__ t, const float* __restrict__ g,
                                    ushort_t* __restrict__ X) {
    int idx = blockIdx.x * blockDim.x + threadIdx.x;      // n*F + i*6 + m
    if (idx >= N_TOK * F_FLAT) return;
    int n = idx / F_FLAT;
    int f = idx - n * F_FLAT;
    int i = f / M_DIM;
    int m = f - i * M_DIM;
    const float* tp = t + (size_t)n * (H_HEAD * M_DIM) + m;
    const float* gp = g + (size_t)n * F_FLAT + i;
    float s = 0.0f;
    #pragma unroll
    for (int h = 0; h < H_HEAD; ++h) s += tp[h * M_DIM] * gp[h * E_EXP];
    X[idx] = (ushort_t)f2bf(s);
}

// bt_bf[d,i] = bf16( b[i,d] )   (512x512 transpose + convert)
__global__ void transpose512_bf_kernel(const float* __restrict__ b, ushort_t* __restrict__ bt) {
    int idx = blockIdx.x * blockDim.x + threadIdx.x;
    if (idx >= D_HALF * E_EXP) return;
    int d = idx >> 9;
    int i = idx & 511;
    bt[idx] = (ushort_t)f2bf(b[(size_t)i * D_HALF + d]);
}

// ---------------------------------------------------------------------------
extern "C" void kernel_launch(void* const* d_in, const int* in_sizes, int n_in,
                              void* d_out, int out_size, void* d_ws, size_t ws_size,
                              hipStream_t stream) {
    const float* x   = (const float*)d_in[0];
    const float* g1  = (const float*)d_in[1];
    const float* g2  = (const float*)d_in[2];
    const float* u1w = (const float*)d_in[3];
    const float* u1b = (const float*)d_in[4];
    const float* u2w = (const float*)d_in[5];
    const float* u2b = (const float*)d_in[6];
    const float* v11 = (const float*)d_in[7];
    const float* v12 = (const float*)d_in[8];
    const float* v21 = (const float*)d_in[9];
    const float* v22 = (const float*)d_in[10];
    const float* b1  = (const float*)d_in[11];
    const float* b2  = (const float*)d_in[12];
    float* out = (float*)d_out;

    // ---- workspace layout (bf16 copies + small f32 buffers) ----------------
    char* wsb = (char*)d_ws;
    size_t o = 0;
    auto alloc = [&](size_t bytes) { char* p = wsb + o; o += (bytes + 255) & ~size_t(255); return p; };

    float*    t1    = (float*)   alloc((size_t)N_TOK * H_HEAD * M_DIM * 4);
    float*    t2    = (float*)   alloc((size_t)N_TOK * H_HEAD * M_DIM * 4);
    ushort_t* x_bf  = (ushort_t*)alloc((size_t)N_TOK * D_HID * 2);
    ushort_t* u1_bf = (ushort_t*)alloc((size_t)F_FLAT * D_HID * 2);
    ushort_t* u2_bf = (ushort_t*)alloc((size_t)F_FLAT * D_HID * 2);
    ushort_t* v11_bf= (ushort_t*)alloc((size_t)D_HALF * F_FLAT * 2);
    ushort_t* v12_bf= (ushort_t*)alloc((size_t)D_HALF * F_FLAT * 2);
    ushort_t* v21_bf= (ushort_t*)alloc((size_t)D_HALF * F_FLAT * 2);
    ushort_t* v22_bf= (ushort_t*)alloc((size_t)D_HALF * F_FLAT * 2);
    ushort_t* b1t_bf= (ushort_t*)alloc((size_t)D_HALF * E_EXP * 2);
    ushort_t* b2t_bf= (ushort_t*)alloc((size_t)D_HALF * E_EXP * 2);
    ushort_t* G1_bf = (ushort_t*)alloc((size_t)N_TOK * E_EXP * 2);
    ushort_t* G2_bf = (ushort_t*)alloc((size_t)N_TOK * E_EXP * 2);
    ushort_t* Y1_bf = (ushort_t*)alloc((size_t)N_TOK * F_FLAT * 2);  // also gs1, then cross12
    ushort_t* Y2_bf = (ushort_t*)alloc((size_t)N_TOK * F_FLAT * 2);  // also gs2, then cross21

    const dim3 blk(256);
    auto cvt = [&](const float* s, ushort_t* d, size_t n) {
        int n8 = (int)(n / 8);
        f32_to_bf16_kernel<<<(n8 + 255) / 256, blk, 0, stream>>>(s, d, n8);
    };

    // 1) one-time bf16 conversions of all GEMM operands
    cvt(x,   x_bf,  (size_t)N_TOK * D_HID);
    cvt(u1w, u1_bf, (size_t)F_FLAT * D_HID);
    cvt(u2w, u2_bf, (size_t)F_FLAT * D_HID);
    cvt(v11, v11_bf, (size_t)D_HALF * F_FLAT);
    cvt(v12, v12_bf, (size_t)D_HALF * F_FLAT);
    cvt(v21, v21_bf, (size_t)D_HALF * F_FLAT);
    cvt(v22, v22_bf, (size_t)D_HALF * F_FLAT);
    transpose512_bf_kernel<<<(D_HALF * E_EXP + 255) / 256, blk, 0, stream>>>(b1, b1t_bf);
    transpose512_bf_kernel<<<(D_HALF * E_EXP + 255) / 256, blk, 0, stream>>>(b2, b2t_bf);
    gate_sum_kernel<<<(N_TOK * E_EXP + 255) / 256, blk, 0, stream>>>(g1, G1_bf);
    gate_sum_kernel<<<(N_TOK * E_EXP + 255) / 256, blk, 0, stream>>>(g2, G2_bf);

    // 2) up-projections, fused bias + relu^2, bf16 output
    dim3 gUp(F_FLAT / 128, N_TOK / 128);                  // (24, 64)
    gemm_bf16_wmma<true,  false, true ><<<gUp, blk, 0, stream>>>(x_bf, D_HID, u1_bf, D_HID, u1b, Y1_bf, F_FLAT, 0, D_HID);
    gemm_bf16_wmma<true,  false, true ><<<gUp, blk, 0, stream>>>(x_bf, D_HID, u2_bf, D_HID, u2b, Y2_bf, F_FLAT, 0, D_HID);

    // 3) per-head mixes t = einsum('bshi,bsim->bshm')
    int nT = N_TOK * H_HEAD * M_DIM;
    head_mix_kernel<<<(nT + 255) / 256, blk, 0, stream>>>(g1, Y1_bf, t1);
    head_mix_kernel<<<(nT + 255) / 256, blk, 0, stream>>>(g2, Y2_bf, t2);

    // 4) gs = G_sum ⊙ Y (in place, bf16)
    int nF = N_TOK * F_FLAT;
    gate_scale_kernel<<<(nF + 255) / 256, blk, 0, stream>>>(Y1_bf, G1_bf);
    gate_scale_kernel<<<(nF + 255) / 256, blk, 0, stream>>>(Y2_bf, G2_bf);

    // 5) main down-projections (initialize both output halves, f32 out)
    dim3 gDn(D_HALF / 128, N_TOK / 128);                  // (4, 64)
    gemm_bf16_wmma<false, false, false><<<gDn, blk, 0, stream>>>(Y1_bf, F_FLAT, v11_bf, F_FLAT, nullptr, out, D_HID, 0,      F_FLAT);
    gemm_bf16_wmma<false, false, false><<<gDn, blk, 0, stream>>>(Y2_bf, F_FLAT, v22_bf, F_FLAT, nullptr, out, D_HID, D_HALF, F_FLAT);

    // 6) cross terms overwrite Y buffers; accumulate their down-projections
    cross_expand_kernel<<<(nF + 255) / 256, blk, 0, stream>>>(t2, g1, Y1_bf);   // x12 pre-proj
    cross_expand_kernel<<<(nF + 255) / 256, blk, 0, stream>>>(t1, g2, Y2_bf);   // x21 pre-proj
    gemm_bf16_wmma<false, true,  false><<<gDn, blk, 0, stream>>>(Y1_bf, F_FLAT, v12_bf, F_FLAT, nullptr, out, D_HID, 0,      F_FLAT);
    gemm_bf16_wmma<false, true,  false><<<gDn, blk, 0, stream>>>(Y2_bf, F_FLAT, v21_bf, F_FLAT, nullptr, out, D_HID, D_HALF, F_FLAT);

    // 7) gate-bias terms: out += G_sum @ b^T
    gemm_bf16_wmma<false, true,  false><<<gDn, blk, 0, stream>>>(G1_bf, E_EXP, b1t_bf, E_EXP, nullptr, out, D_HID, 0,      E_EXP);
    gemm_bf16_wmma<false, true,  false><<<gDn, blk, 0, stream>>>(G2_bf, E_EXP, b2t_bf, E_EXP, nullptr, out, D_HID, D_HALF, E_EXP);
}
